// CrossNetV2_38611755991213
// MI455X (gfx1250) — matrix operand order repaired and checked by reading, not compiled, not run on hardware
//
#include <hip/hip_runtime.h>
#include <hip/hip_bf16.h>

// ---------------------------------------------------------------------------
// CrossNetV2 low-rank MoE, fused per-row-strip pipeline on gfx1250 (wave32).
// bf16 WMMA (v_wmma_f32_16x16x32_bf16) fed from LDS weight panels that are
// double-buffered via GLOBAL_LOAD_ASYNC_TO_LDS_B128 (ASYNCcnt) prefetch.
// B-fragment feed is explicitly software-pipelined (5-deep rotation, PF=4)
// so partial s_wait_dscnt waits survive the 256-VGPR cap.
// ---------------------------------------------------------------------------

typedef __bf16        v16bf __attribute__((ext_vector_type(16)));
typedef __bf16        v4bf  __attribute__((ext_vector_type(4)));
typedef float         v8f   __attribute__((ext_vector_type(8)));
typedef float         v4f   __attribute__((ext_vector_type(4)));
typedef unsigned int  v4u   __attribute__((ext_vector_type(4)));

union Frag {            // 16 bf16 = 8 VGPRs, loaded as two 128-bit pieces
    v16bf bf;
    v4u   u2[2];
};

#define CN_B   16384
#define CN_D   1024
#define CN_R   64
#define CN_E   4
#define CN_L   3
#define CN_N1  272          // 256 expert cols + 4 gate cols + 12 zero pad
#define CN_RT  64           // rows per workgroup (4 waves x 16-row strips)
#define CN_XP  1032         // LDS pitch (elems) for x tile    (2064B)
#define CN_HP  264          // LDS pitch (elems) for hg tile   (528B)
#define CN_UP  72           // LDS pitch (elems) for U panel   (144B)
#define CN_KB  64           // K-depth per staged U panel  -> 16 panels
#define CN_NB  128          // N-cols per staged V panel   -> 8 panels

// LDS element offsets
#define OFF_XS   0
#define OFF_HG   (CN_RT * CN_XP)                       // 66048
#define OFF_PAN0 (OFF_HG + CN_RT * CN_HP)              // 82944
#define PAN_ELEMS (CN_NB * CN_HP)                      // 33792 elems = 67584 B per buffer
#define OFF_PAN1 (OFF_PAN0 + PAN_ELEMS)
#define LDS_ELEMS (OFF_PAN1 + PAN_ELEMS)               // 150528 elems = 301056 B

static __device__ inline v8f vzero8() {
    v8f z;
#pragma unroll
    for (int i = 0; i < 8; ++i) z[i] = 0.0f;
    return z;
}

// One 16-byte lane-chunk, global -> LDS, tracked by ASYNCcnt (no VGPR data).
static __device__ inline void async_cp16(const __bf16* gsrc, unsigned lds) {
    asm volatile("global_load_async_to_lds_b128 %0, %1, off"
                 :: "v"(lds), "v"(gsrc) : "memory");
}

static __device__ inline void wait_async0() {
#if __has_builtin(__builtin_amdgcn_s_wait_asynccnt)
    __builtin_amdgcn_s_wait_asynccnt(0);
#else
    asm volatile("s_wait_asynccnt 0x0" ::: "memory");
#endif
}

// Stage U k-panel: 272 rows x 64 K (bf16) -> LDS pitch 72. 2176 16B chunks.
static __device__ inline void stage_U(unsigned dstLds, const __bf16* Ul, int kb0, int t) {
#pragma unroll
    for (int i = 0; i < 17; ++i) {
        int c = i * 128 + t;            // 0..2175
        int n = c >> 3;                 // row (column of GEMM1)
        int j = (c & 7) * 8;            // elem offset within 64-K row
        async_cp16(Ul + (size_t)n * CN_D + kb0 + j,
                   dstLds + (unsigned)(n * CN_UP + j) * 2u);
    }
}

// Stage V n-panel: 128 cols x 256 K (bf16) -> LDS pitch 264. 4096 16B chunks.
static __device__ inline void stage_V(unsigned dstLds, const __bf16* Vl, int ntb, int t) {
#pragma unroll
    for (int i = 0; i < 32; ++i) {
        int c = i * 128 + t;            // 0..4095
        int col = c >> 5;               // panel-local output column
        int j = (c & 31) * 8;           // elem offset within 256-K row
        async_cp16(Vl + (size_t)(ntb * CN_NB + col) * 256 + j,
                   dstLds + (unsigned)(col * CN_HP + j) * 2u);
    }
}

static __device__ inline void load_frag(Frag& f, const __bf16* p) {
    f.u2[0] = *(const v4u*)(p);
    f.u2[1] = *(const v4u*)(p + 16);
}

// ---- weight pre-pack: U/gW -> Ubf[l][n=272][d=1024], bf16, K(d)-contiguous --
__global__ __launch_bounds__(256) void pack_U(const float* __restrict__ U,
                                              const float* __restrict__ gW,
                                              __bf16* __restrict__ Ubf) {
    int idx = blockIdx.x * 256 + threadIdx.x;       // over 3*272*1024
    if (idx >= CN_L * CN_N1 * CN_D) return;
    int d = idx & (CN_D - 1);
    int n = (idx >> 10) % CN_N1;
    int l = idx / (CN_N1 * CN_D);
    float v = 0.0f;
    if (n < 256) {                                   // expert col n=(e,r)
        int e = n >> 6, r = n & 63;
        v = U[((l * CN_E + e) * CN_D + d) * CN_R + r];
    } else if (n < 260) {                            // gate col
        int e = n - 256;
        v = gW[(l * CN_D + d) * CN_E + e];
    }
    Ubf[idx] = (__bf16)v;
}

// ---- weight pre-pack: V -> Vbf[l][d=1024][n=256], bf16, K(n)-contiguous ----
__global__ __launch_bounds__(256) void pack_V(const float* __restrict__ V,
                                              __bf16* __restrict__ Vbf) {
    int idx = blockIdx.x * 256 + threadIdx.x;       // over 3*1024*256
    if (idx >= CN_L * CN_D * 256) return;
    int n = idx & 255;
    int d = (idx >> 8) & (CN_D - 1);
    int l = idx >> 18;
    int e = n >> 6, r = n & 63;
    Vbf[idx] = (__bf16)V[((l * CN_E + e) * CN_R + r) * CN_D + d];
}

// ---------------------------- main fused kernel -----------------------------
__global__ __launch_bounds__(128)
__attribute__((amdgpu_num_vgpr(256), amdgpu_waves_per_eu(4)))
void crossnet_fused(
    const float*  __restrict__ x0g,
    const __bf16* __restrict__ Ubf,
    const __bf16* __restrict__ Vbf,
    const float*  __restrict__ gb,
    const float*  __restrict__ bias,
    float*        __restrict__ out)
{
    extern __shared__ __bf16 smem[];
    __bf16* xs  = smem + OFF_XS;      // [64][1032] running x (bf16)
    __bf16* hgs = smem + OFF_HG;      // [64][264]  gated hidden

    const unsigned ldsBase = (unsigned)(size_t)(void*)smem;
    const unsigned panLds0 = ldsBase + OFF_PAN0 * 2u;

    const int t      = threadIdx.x;
    const int wave   = t >> 5;
    const int lane   = t & 31;
    const int lm     = lane & 15;             // col (B/C) or row (A) within tile
    const int lh     = (lane >> 4) & 1;       // lane half
    const int strip  = wave * 16;             // this wave's 16-row strip
    const int rowBlk = blockIdx.x * CN_RT;    // global row base

    // ---- cooperative load of x0 tile -> bf16 xs ----
#pragma unroll 4
    for (int i = 0; i < 128; ++i) {
        int q  = i * 128 + t;                 // float4 index, 0..16383
        int r  = q >> 8;                      // 256 float4 per row
        int c4 = (q & 255) * 4;
        v4f v = *(const v4f*)(x0g + (size_t)(rowBlk + r) * CN_D + c4);
        v4bf bv;
#pragma unroll
        for (int j = 0; j < 4; ++j) bv[j] = (__bf16)v[j];
        *(v4bf*)(xs + r * CN_XP + c4) = bv;
    }
    __syncthreads();

    for (int l = 0; l < CN_L; ++l) {
        const __bf16* Ul = Ubf + (size_t)l * CN_N1 * CN_D;
        const __bf16* Vl = Vbf + (size_t)l * CN_D * 256;
        int buf = 0;

        // ============ GEMM1: Y[16x272] = x_strip[16x1024] @ [U | gW] ========
        v8f acc[17];
#pragma unroll
        for (int n = 0; n < 17; ++n) acc[n] = vzero8();

        const __bf16* arow = xs + (strip + lm) * CN_XP + lh * 8;

        stage_U(panLds0, Ul, 0, t);
        wait_async0();
        __syncthreads();

        for (int kb = 0; kb < CN_D / CN_KB; ++kb) {
            if (kb + 1 < CN_D / CN_KB)
                stage_U(panLds0 + (unsigned)(buf ^ 1) * (PAN_ELEMS * 2u),
                        Ul, (kb + 1) * CN_KB, t);                 // async prefetch
            const __bf16* Up   = smem + OFF_PAN0 + buf * PAN_ELEMS;
            const __bf16* Bbas = Up + lm * CN_UP + lh * 8;

            // both A fragments for this kb (K = kb*64 .. +63)
            Frag a[2];
            a[0].u2[0] = *(const v4u*)(arow + kb * CN_KB);
            a[0].u2[1] = *(const v4u*)(arow + kb * CN_KB + 16);
            a[1].u2[0] = *(const v4u*)(arow + kb * CN_KB + 32);
            a[1].u2[1] = *(const v4u*)(arow + kb * CN_KB + 48);

            // flattened (ks, n) stream: m = ks*17 + n, software-pipelined
            Frag b[5];
#pragma unroll
            for (int m = 0; m < 4; ++m) {
                const int ks = m / 17, n = m % 17;
                load_frag(b[m % 5], Bbas + n * (16 * CN_UP) + ks * 32);
            }
#pragma unroll
            for (int m = 0; m < 34; ++m) {
                if (m + 4 < 34) {
                    const int ks = (m + 4) / 17, n = (m + 4) % 17;
                    load_frag(b[(m + 4) % 5], Bbas + n * (16 * CN_UP) + ks * 32);
                }
                const int ks = m / 17, n = m % 17;
                acc[n] = __builtin_amdgcn_wmma_f32_16x16x32_bf16(
                    false, a[ks].bf, false, b[m % 5].bf, (short)0, acc[n], false, false);
            }

            wait_async0();          // prefetch done (issued before compute)
            __syncthreads();        // everyone finished reading pan[buf]
            buf ^= 1;
        }

        // ============ in-wave softmax over E=4 (gate tile = acc[16]) ========
        float g0 = gb[l * CN_E + 0], g1 = gb[l * CN_E + 1];
        float g2 = gb[l * CN_E + 2], g3 = gb[l * CN_E + 3];
        const int base = lane & 16;
        float gate[4][8];
#pragma unroll
        for (int v = 0; v < 8; ++v) {
            float s = acc[16][v];
            float l0 = __shfl(s, base + 0, 32) + g0;
            float l1 = __shfl(s, base + 1, 32) + g1;
            float l2 = __shfl(s, base + 2, 32) + g2;
            float l3 = __shfl(s, base + 3, 32) + g3;
            float m  = fmaxf(fmaxf(l0, l1), fmaxf(l2, l3));
            float e0 = __expf(l0 - m), e1 = __expf(l1 - m);
            float e2 = __expf(l2 - m), e3 = __expf(l3 - m);
            float inv = 1.0f / (e0 + e1 + e2 + e3);
            gate[0][v] = e0 * inv; gate[1][v] = e1 * inv;
            gate[2][v] = e2 * inv; gate[3][v] = e3 * inv;
        }

        // ============ hg = relu(h) * gate  -> LDS (row-major bf16) ==========
#pragma unroll
        for (int n = 0; n < 16; ++n) {
            const int e = n >> 2;                       // 64 cols per expert
            __bf16* dst = hgs + (strip + lh * 8) * CN_HP + n * 16 + lm;
#pragma unroll
            for (int v = 0; v < 8; ++v) {
                float h = acc[n][v];
                h = h > 0.0f ? h * gate[e][v] : 0.0f;
                dst[v * CN_HP] = (__bf16)h;
            }
        }

        // ============ GEMM2: mixed[16x1024] = hg[16x256] @ V2[256x1024] =====
        stage_V(panLds0 + (unsigned)buf * (PAN_ELEMS * 2u), Vl, 0, t);

        // A fragments (K=256) preloaded once, reused across all 64 N tiles
        Frag a2[8];
        const __bf16* hrow = hgs + (strip + lm) * CN_HP + lh * 8;
#pragma unroll
        for (int kk = 0; kk < 8; ++kk) {
            a2[kk].u2[0] = *(const v4u*)(hrow + kk * 32);
            a2[kk].u2[1] = *(const v4u*)(hrow + kk * 32 + 16);
        }

        wait_async0();
        __syncthreads();

        for (int ntb = 0; ntb < 1024 / (16 * 8); ++ntb) {     // 8 panels
            if (ntb + 1 < 8)
                stage_V(panLds0 + (unsigned)(buf ^ 1) * (PAN_ELEMS * 2u),
                        Vl, ntb + 1, t);                      // async prefetch
            const __bf16* Vp   = smem + OFF_PAN0 + buf * PAN_ELEMS;
            const __bf16* Bbas = Vp + lm * CN_HP + lh * 8;

            // flattened (ntl, kk) stream: m = ntl*8 + kk, software-pipelined
            Frag b[5];
#pragma unroll
            for (int m = 0; m < 4; ++m)
                load_frag(b[m % 5], Bbas + (m >> 3) * (16 * CN_HP) + (m & 7) * 32);

            v8f macc = vzero8();
#pragma unroll
            for (int m = 0; m < 64; ++m) {
                if (m + 4 < 64) {
                    const int mp = m + 4;
                    load_frag(b[mp % 5], Bbas + (mp >> 3) * (16 * CN_HP) + (mp & 7) * 32);
                }
                macc = __builtin_amdgcn_wmma_f32_16x16x32_bf16(
                    false, a2[m & 7].bf, false, b[m % 5].bf, (short)0, macc, false, false);

                if ((m & 7) == 7) {
                    // epilogue: x = x0*mixed + bias + x (x0 re-read: L2-resident)
                    const int nt  = ntb * 8 + (m >> 3);
                    const int col = nt * 16 + lm;
                    const float bv = bias[l * CN_D + col];
#pragma unroll
                    for (int v = 0; v < 8; ++v) {
                        const int r  = strip + lh * 8 + v;
                        const int xo = r * CN_XP + col;
                        float x0v = x0g[(size_t)(rowBlk + r) * CN_D + col];
                        float xn  = x0v * macc[v] + bv + (float)xs[xo];
                        xs[xo] = (__bf16)xn;
                        if (l == CN_L - 1)
                            out[(size_t)(rowBlk + r) * CN_D + col] = xn;
                    }
                    macc = vzero8();
                }
            }

            wait_async0();
            __syncthreads();
            buf ^= 1;
        }
    }
}

// ---------------------------------------------------------------------------
extern "C" void kernel_launch(void* const* d_in, const int* in_sizes, int n_in,
                              void* d_out, int out_size, void* d_ws, size_t ws_size,
                              hipStream_t stream)
{
    (void)in_sizes; (void)n_in; (void)out_size; (void)ws_size;
    const float* x0   = (const float*)d_in[0];
    const float* U    = (const float*)d_in[1];
    const float* V    = (const float*)d_in[2];
    const float* gW   = (const float*)d_in[3];
    const float* gb   = (const float*)d_in[4];
    const float* bias = (const float*)d_in[5];
    float* out = (float*)d_out;

    __bf16* Ubf = (__bf16*)d_ws;                                             // 3*272*1024
    __bf16* Vbf = (__bf16*)((char*)d_ws + (size_t)CN_L * CN_N1 * CN_D * 2);  // 3*1024*256

    const int nU = CN_L * CN_N1 * CN_D;
    const int nV = CN_L * CN_D * 256;
    pack_U<<<(nU + 255) / 256, 256, 0, stream>>>(U, gW, Ubf);
    pack_V<<<(nV + 255) / 256, 256, 0, stream>>>(V, Vbf);

    const size_t ldsBytes = (size_t)LDS_ELEMS * sizeof(__bf16);              // 301056 B
    crossnet_fused<<<CN_B / CN_RT, 128, ldsBytes, stream>>>(x0, Ubf, Vbf, gb, bias, out);
}
